// LSTMForecaster_78176994722526
// MI455X (gfx1250) — matrix operand order repaired and checked by reading, not compiled, not run on hardware
//
#include <hip/hip_runtime.h>

#define B_SZ   1024
#define T_SZ   512
#define DIN    18
#define H_SZ   128
#define DOUT   24

typedef __attribute__((ext_vector_type(16))) __bf16 v16bf;
typedef __attribute__((ext_vector_type(8)))  __bf16 bf16x8;
typedef __attribute__((ext_vector_type(8)))  float  v8f;

// ---- workspace layout (bf16 elements) ----
#define WIH0_OFF 0                 // [512][32]  (K padded 18->32 with zeros)
#define WHH0_OFF 16384             // [512][128]
#define WIH1_OFF 81920             // [512][128]
#define WHH1_OFF 147456            // [512][128]
#define WS_ELEMS 212992

// ---- LDS strides (bf16 elements); +8 pad rotates banks (272B rows) ----
#define WSTR 136
#define HSTR 136
#define XSTR 40
#define SMEM_ELEMS (2 * 512 * WSTR + 2 * 16 * HSTR + 16 * XSTR)   // 144256
#define SMEM_BYTES (SMEM_ELEMS * 2)                               // 288512 < 320KB

// ------------------------------------------------------------------
// helpers
// ------------------------------------------------------------------
__device__ __forceinline__ v16bf cat8(bf16x8 lo, bf16x8 hi) {
  return __builtin_shufflevector(lo, hi, 0,1,2,3,4,5,6,7,8,9,10,11,12,13,14,15);
}

// A fragment (16x32, M=row): lanes 0-15 hold K=kb+[0..7] (V0-3) and kb+16+[0..7] (V4-7),
// lanes 16-31 hold kb+8+[0..7] and kb+24+[0..7].
__device__ __forceinline__ v16bf load_frag_A(const __bf16* p, int rowStride, int row,
                                             int kb, int half) {
  const __bf16* base = p + row * rowStride + kb + 8 * half;
  bf16x8 lo = *(const bf16x8*)(base);
  bf16x8 hi = *(const bf16x8*)(base + 16);
  return cat8(lo, hi);
}

// B fragment (32x16, lane%16 = N col; lanes 0-15 hold K=kb+[0..7],kb+16+[0..7]; lanes 16-31 K=kb+8+[0..7],kb+24+[0..7]).
// W is row-major [N][K] (gates = h @ W^T), so this is two 16-byte chunks.
__device__ __forceinline__ v16bf load_frag_B(const __bf16* p, int rowStride, int nrow,
                                             int kb, int half) {
  const __bf16* base = p + nrow * rowStride + kb + 8 * half;
  bf16x8 lo = *(const bf16x8*)(base);
  bf16x8 hi = *(const bf16x8*)(base + 16);
  return cat8(lo, hi);
}

__device__ __forceinline__ v8f wmma_bf16(v16bf a, v16bf b, v8f c) {
  return __builtin_amdgcn_wmma_f32_16x16x32_bf16(false, a, false, b, (short)0, c,
                                                 false, false);
}

__device__ __forceinline__ v8f vsplat(float x) {
  v8f r;
#pragma unroll
  for (int e = 0; e < 8; ++e) r[e] = x;
  return r;
}

__device__ __forceinline__ float sigf(float x) {
  float e = __expf(-x);
  return __builtin_amdgcn_rcpf(1.0f + e);
}
__device__ __forceinline__ float tanh_fast(float x) {
  float e = __expf(2.0f * x);
  return (e - 1.0f) * __builtin_amdgcn_rcpf(e + 1.0f);
}

// ------------------------------------------------------------------
// prep kernel: pack all weight matrices to bf16 in workspace
// ------------------------------------------------------------------
__global__ void pack_weights(const float* __restrict__ Wih0,  // [512][18]
                             const float* __restrict__ Whh0,  // [512][128]
                             const float* __restrict__ Wih1,  // [512][128]
                             const float* __restrict__ Whh1,  // [512][128]
                             __bf16* __restrict__ ws) {
  int i = blockIdx.x * 256 + threadIdx.x;
  if (i >= WS_ELEMS) return;
  if (i < WHH0_OFF) {                       // Wih0 padded [512][32]
    int n = i >> 5, k = i & 31;
    ws[i] = (k < DIN) ? (__bf16)Wih0[n * DIN + k] : (__bf16)0.0f;
  } else if (i < WIH1_OFF) {
    ws[i] = (__bf16)Whh0[i - WHH0_OFF];
  } else if (i < WHH1_OFF) {
    ws[i] = (__bf16)Wih1[i - WIH1_OFF];
  } else {
    ws[i] = (__bf16)Whh1[i - WHH1_OFF];
  }
}

// ------------------------------------------------------------------
// fused 2-layer LSTM + linear head; one WG owns 16 batch rows for all T
// ------------------------------------------------------------------
__global__ __launch_bounds__(256, 1)
void lstm_fused(const float* __restrict__ xp,        // [B][T][18]
                const __bf16* __restrict__ wsbf,
                const float* __restrict__ b_ih0, const float* __restrict__ b_hh0,
                const float* __restrict__ b_ih1, const float* __restrict__ b_hh1,
                const float* __restrict__ W_lin,     // [24][128]
                const float* __restrict__ b_lin,     // [24]
                float* __restrict__ out)             // [B][24]
{
  extern __shared__ __bf16 smem[];
  __bf16* ldsWhh0 = smem;                       // [512][WSTR]
  __bf16* ldsWhh1 = smem + 512 * WSTR;          // [512][WSTR]
  __bf16* h0buf   = smem + 2 * 512 * WSTR;      // [16][HSTR]
  __bf16* h1buf   = h0buf + 16 * HSTR;          // [16][HSTR]
  __bf16* xbuf    = h1buf + 16 * HSTR;          // [16][XSTR]

  const int tid   = threadIdx.x;
  const int lane  = tid & 31;
  const int wv    = tid >> 5;        // wave 0..7 -> hidden slice [16*wv, 16*wv+16)
  const int nloc  = lane & 15;
  const int half  = lane >> 4;
  const int bBase = blockIdx.x * 16;

  const __bf16* Wih0p = wsbf + WIH0_OFF;
  const __bf16* Whh0p = wsbf + WHH0_OFF;
  const __bf16* Wih1p = wsbf + WIH1_OFF;
  const __bf16* Whh1p = wsbf + WHH1_OFF;

  // ---- fill LDS with recurrent weights (bf16, padded stride) ----
  for (int s = tid; s < 512 * 16; s += 256) {
    int r = s >> 4, ch = (s & 15) * 8;
    *(uint4*)(ldsWhh0 + r * WSTR + ch) = *(const uint4*)(Whh0p + r * 128 + ch);
    *(uint4*)(ldsWhh1 + r * WSTR + ch) = *(const uint4*)(Whh1p + r * 128 + ch);
  }
  for (int s = tid; s < 16 * HSTR; s += 256) {
    h0buf[s] = (__bf16)0.0f;
    h1buf[s] = (__bf16)0.0f;
  }

  // ---- per-lane fused biases and register-resident W_ih0 B-fragments ----
  float b0g[4], b1g[4];
  v16bf wih0f[4];
#pragma unroll
  for (int g = 0; g < 4; ++g) {
    int n   = g * H_SZ + 16 * wv + nloc;
    b0g[g]  = b_ih0[n] + b_hh0[n];
    b1g[g]  = b_ih1[n] + b_hh1[n];
    wih0f[g] = load_frag_B(Wih0p, 32, n, 0, half);
  }

  v8f c0 = vsplat(0.0f), c1 = vsplat(0.0f);
  __syncthreads();

  for (int t = 0; t < T_SZ; ++t) {
    // ---- stage x(t) tile as padded bf16 [16][32] ----
    for (int s = tid; s < 16 * 32; s += 256) {
      int r = s >> 5, k = s & 31;
      float xv = (k < DIN) ? xp[((size_t)(bBase + r) * T_SZ + t) * DIN + k] : 0.0f;
      xbuf[r * XSTR + k] = (__bf16)xv;
    }
    if (t + 1 < T_SZ && tid < 16)  // pull next step's x toward the WGP
      __builtin_prefetch(xp + ((size_t)(bBase + tid) * T_SZ + (t + 1)) * DIN, 0, 1);
    __syncthreads();  // (2) xbuf ready; prev h writes visible

    // ---- A fragments (h0(t-1), h1(t-1), x(t)) ----
    v16bf ax = load_frag_A(xbuf, XSTR, nloc, 0, half);
    v16bf ah0[4], ah1[4];
#pragma unroll
    for (int kk = 0; kk < 4; ++kk) {
      ah0[kk] = load_frag_A(h0buf, HSTR, nloc, 32 * kk, half);
      ah1[kk] = load_frag_A(h1buf, HSTR, nloc, 32 * kk, half);
    }

    // ---- layer0 gates: x @ Wih0^T + h0 @ Whh0^T + b ----
    v8f acc0[4];
#pragma unroll
    for (int g = 0; g < 4; ++g) {
      int n = g * H_SZ + 16 * wv + nloc;
      v8f a = vsplat(b0g[g]);
      a = wmma_bf16(ax, wih0f[g], a);
#pragma unroll
      for (int kk = 0; kk < 4; ++kk)
        a = wmma_bf16(ah0[kk], load_frag_B(ldsWhh0, WSTR, n, 32 * kk, half), a);
      acc0[g] = a;
    }

    // ---- layer1 partial: b + h1(t-1) @ Whh1^T (reads old h1buf) ----
    v8f acc1[4];
#pragma unroll
    for (int g = 0; g < 4; ++g) {
      int n = g * H_SZ + 16 * wv + nloc;
      v8f a = vsplat(b1g[g]);
#pragma unroll
      for (int kk = 0; kk < 4; ++kk)
        a = wmma_bf16(ah1[kk], load_frag_B(ldsWhh1, WSTR, n, 32 * kk, half), a);
      acc1[g] = a;
    }

    // ---- layer0 cell update (c in registers, identical C-layout tiles) ----
    v8f h0n;
#pragma unroll
    for (int e = 0; e < 8; ++e) {
      float ig = sigf(acc0[0][e]);
      float fg = sigf(acc0[1][e]);
      float gg = tanh_fast(acc0[2][e]);
      float og = sigf(acc0[3][e]);
      c0[e]  = fg * c0[e] + ig * gg;
      h0n[e] = og * tanh_fast(c0[e]);
    }

    __syncthreads();  // (4) all reads of h0buf/h1buf done
#pragma unroll
    for (int v = 0; v < 8; ++v)  // C-layout: (m = 8*half+v, n = nloc) -> lane-transpose via LDS
      h0buf[(8 * half + v) * HSTR + 16 * wv + nloc] = (__bf16)h0n[v];
    __syncthreads();  // (6) h0buf = h0(t)

    // ---- layer1 finish: + h0(t) @ Wih1^T (Wih1 streamed, L2-resident) ----
#pragma unroll
    for (int kk = 0; kk < 4; ++kk) {
      v16bf a = load_frag_A(h0buf, HSTR, nloc, 32 * kk, half);
#pragma unroll
      for (int g = 0; g < 4; ++g) {
        int n = g * H_SZ + 16 * wv + nloc;
        acc1[g] = wmma_bf16(a, load_frag_B(Wih1p, 128, n, 32 * kk, half), acc1[g]);
      }
    }

    // ---- layer1 cell update ----
    v8f h1n;
#pragma unroll
    for (int e = 0; e < 8; ++e) {
      float ig = sigf(acc1[0][e]);
      float fg = sigf(acc1[1][e]);
      float gg = tanh_fast(acc1[2][e]);
      float og = sigf(acc1[3][e]);
      c1[e]  = fg * c1[e] + ig * gg;
      h1n[e] = og * tanh_fast(c1[e]);
    }
#pragma unroll
    for (int v = 0; v < 8; ++v)
      h1buf[(8 * half + v) * HSTR + 16 * wv + nloc] = (__bf16)h1n[v];
    // next iteration's barrier (2) orders these writes vs. reads
  }

  __syncthreads();
  // ---- linear head on h1(T-1): [16][128] @ [24][128]^T ----
  for (int idx = tid; idx < 16 * DOUT; idx += 256) {
    int r = idx / DOUT, j = idx % DOUT;
    float s = b_lin[j];
#pragma unroll 8
    for (int k = 0; k < H_SZ; ++k)
      s += (float)h1buf[r * HSTR + k] * W_lin[j * H_SZ + k];
    out[(size_t)(bBase + r) * DOUT + j] = s;
  }
}

// ------------------------------------------------------------------
extern "C" void kernel_launch(void* const* d_in, const int* in_sizes, int n_in,
                              void* d_out, int out_size, void* d_ws, size_t ws_size,
                              hipStream_t stream) {
  (void)in_sizes; (void)n_in; (void)out_size; (void)ws_size;
  const float* xp   = (const float*)d_in[0];
  const float* Wih0 = (const float*)d_in[1];
  const float* Whh0 = (const float*)d_in[2];
  const float* bih0 = (const float*)d_in[3];
  const float* bhh0 = (const float*)d_in[4];
  const float* Wih1 = (const float*)d_in[5];
  const float* Whh1 = (const float*)d_in[6];
  const float* bih1 = (const float*)d_in[7];
  const float* bhh1 = (const float*)d_in[8];
  const float* Wlin = (const float*)d_in[9];
  const float* blin = (const float*)d_in[10];
  __bf16* wsbf = (__bf16*)d_ws;
  float* out   = (float*)d_out;

  // allow 288.5 KB dynamic LDS (WGP has 320 KB)
  hipFuncSetAttribute(reinterpret_cast<const void*>(lstm_fused),
                      hipFuncAttributeMaxDynamicSharedMemorySize, SMEM_BYTES);

  pack_weights<<<(WS_ELEMS + 255) / 256, 256, 0, stream>>>(Wih0, Whh0, Wih1, Whh1, wsbf);
  lstm_fused<<<B_SZ / 16, 256, SMEM_BYTES, stream>>>(xp, wsbf, bih0, bhh0, bih1, bhh1,
                                                     Wlin, blin, out);
}
